// LovaszLossSoftmax_48120813584431
// MI455X (gfx1250) — compile-verified, matrix operand order checked by reading, not run on hardware
//
#include <hip/hip_runtime.h>
#include <stdint.h>

#define P_TOT  (1u << 21)   // 8*512*512 pixels
#define HW     (1u << 18)   // 512*512
#define NCLS   19
#define TILE   4096
#define NBLK   512          // P_TOT / TILE

typedef int v8i __attribute__((ext_vector_type(8)));

// ---------------------------------------------------------------- init
__global__ void k_init(unsigned* Gcnt, float* lossArr) {
  int t = threadIdx.x;
  if (t < NCLS) { Gcnt[t] = 0u; lossArr[t] = 0.0f; }
}

// ------------------------------------------------- per-pixel softmax stats
__global__ __launch_bounds__(256) void k_softmax_stats(const float* __restrict__ x,
                                                       float* __restrict__ mArr,
                                                       float* __restrict__ invS) {
  unsigned p  = blockIdx.x * 256u + threadIdx.x;
  unsigned b  = p >> 18;
  unsigned hw = p & (HW - 1u);
  const float* px = x + (size_t)b * NCLS * HW + hw;
  float v[NCLS];
#pragma unroll
  for (int c = 0; c < NCLS; ++c) v[c] = px[(size_t)c * HW];
  float m = v[0];
#pragma unroll
  for (int c = 1; c < NCLS; ++c) m = fmaxf(m, v[c]);
  float s = 0.0f;
#pragma unroll
  for (int c = 0; c < NCLS; ++c) s += __expf(v[c] - m);
  mArr[p] = m;
  invS[p] = 1.0f / s;
}

// ------------------------------------- label histogram via v_wmma_i32_16x16x64_iu8
// A[m][k] = (label[k]==m) one-hot (8-bit), B = ones  =>  D[m][n] = count(label==m)
__global__ __launch_bounds__(256) void k_label_hist(const int* __restrict__ labels,
                                                    unsigned* __restrict__ Gcnt) {
  __shared__ unsigned char lbl[8][64];
  const int t = threadIdx.x, w = t >> 5, lane = t & 31;
  const int waveId = blockIdx.x * 8 + w;           // 128 blocks -> 1024 waves
  v8i accLo = {0, 0, 0, 0, 0, 0, 0, 0};
  v8i accHi = {0, 0, 0, 0, 0, 0, 0, 0};
  v8i bOnes;
#pragma unroll
  for (int i = 0; i < 8; ++i) bOnes[i] = 0x01010101;
  const int m     = lane & 15;
  const int khalf = (lane < 16) ? 0 : 8;
  for (int it = 0; it < 32; ++it) {                // 32*1024*64 = 2^21 labels
    int base = (it * 1024 + waveId) * 64;
    lbl[w][lane]      = (unsigned char)labels[base + lane];
    lbl[w][lane + 32] = (unsigned char)labels[base + lane + 32];
    __syncthreads();
    v8i A0, A1;
#pragma unroll
    for (int vv = 0; vv < 8; ++vv) {
      // 8-bit A 16x64 layout: v0:K0-3/8-11 v1:K4-7/12-15 v2:K16-19/24-27
      // v3:K20-23/28-31 v4..v7: +32   (upper half lanes: +8)
      const int k0 = ((vv >> 1) << 4) + ((vv & 1) << 2) + khalf;
      unsigned a0 = 0u, a1 = 0u;
#pragma unroll
      for (int bb = 0; bb < 4; ++bb) {
        int lb = (int)lbl[w][k0 + bb];
        a0 |= (unsigned)(lb == m)      << (8 * bb);
        a1 |= (unsigned)(lb == m + 16) << (8 * bb);
      }
      A0[vv] = (int)a0; A1[vv] = (int)a1;
    }
    accLo = __builtin_amdgcn_wmma_i32_16x16x64_iu8(false, A0, false, bOnes, accLo, false, false);
    accHi = __builtin_amdgcn_wmma_i32_16x16x64_iu8(false, A1, false, bOnes, accHi, false, false);
    __syncthreads();
  }
  // C/D layout: VGPR j -> lanes0-15: M=j (N=lane), lanes16-31: M=j+8
  if (lane == 0) {
#pragma unroll
    for (int j = 0; j < 8; ++j) atomicAdd(&Gcnt[j], (unsigned)accLo[j]);
#pragma unroll
    for (int j = 0; j < 3; ++j) atomicAdd(&Gcnt[16 + j], (unsigned)accHi[j]);
  }
  if (lane == 16) {
#pragma unroll
    for (int j = 0; j < 8; ++j) atomicAdd(&Gcnt[8 + j], (unsigned)accLo[j]);
  }
}

// ---------------------------------------------------------- per-class keys
__global__ __launch_bounds__(256) void k_build_keys(const float* __restrict__ x,
                                                    const int* __restrict__ labels,
                                                    const float* __restrict__ mArr,
                                                    const float* __restrict__ invS,
                                                    unsigned* __restrict__ keys, int c) {
  unsigned p  = blockIdx.x * 256u + threadIdx.x;
  unsigned b  = p >> 18;
  unsigned hw = p & (HW - 1u);
  float xv   = x[((size_t)b * NCLS + c) * HW + hw];
  float prob = __expf(xv - mArr[p]) * invS[p];
  prob = fminf(fmaxf(prob, 0.0f), 1.0f);
  unsigned fg = (labels[p] == c) ? 1u : 0u;
  float err   = fg ? (1.0f - prob) : prob;
  unsigned bits = __float_as_uint(err);
  keys[p] = ~((bits & 0xFFFFFFFEu) | fg);   // ascending uint == descending error
}

// ----------------------------------------- radix pass 1/3: per-block histogram
// 16 KB tile staged into LDS via async-to-LDS (ASYNCcnt path)
__global__ __launch_bounds__(256) void k_hist(const unsigned* __restrict__ src,
                                              unsigned* __restrict__ hist, int shift) {
  __shared__ unsigned tile[TILE];
  __shared__ unsigned cnt[256];
  const int t = threadIdx.x;
  const unsigned blk = blockIdx.x;
  const unsigned ldsBase = (unsigned)(uintptr_t)(&tile[0]);
#pragma unroll
  for (int j = 0; j < 4; ++j) {
    unsigned byteOff = (unsigned)j * 4096u + (unsigned)t * 16u;
    unsigned l = ldsBase + byteOff;
    unsigned g = blk * 16384u + byteOff;
    asm volatile("global_load_async_to_lds_b128 %0, %1, %2"
                 :: "v"(l), "v"(g), "s"(src) : "memory");
  }
  cnt[t] = 0u;
  asm volatile("s_wait_asynccnt 0" ::: "memory");
  __syncthreads();
#pragma unroll
  for (int i = 0; i < 16; ++i) {
    unsigned key = tile[i * 256 + t];
    atomicAdd(&cnt[(key >> shift) & 255u], 1u);
  }
  __syncthreads();
  hist[(unsigned)t * NBLK + blk] = cnt[t];
}

// ------------------------------ radix pass 2a: scan blocks within each digit
__global__ __launch_bounds__(512) void k_scan_blocks(unsigned* __restrict__ hist,
                                                     unsigned* __restrict__ dtot) {
  __shared__ unsigned buf[512];
  const int d = blockIdx.x, t = threadIdx.x;
  unsigned v = hist[d * NBLK + t];
  buf[t] = v;
  __syncthreads();
  for (int off = 1; off < 512; off <<= 1) {
    unsigned add = (t >= off) ? buf[t - off] : 0u;
    __syncthreads();
    buf[t] += add;
    __syncthreads();
  }
  hist[d * NBLK + t] = buf[t] - v;   // exclusive within digit
  if (t == 511) dtot[d] = buf[511];
}

// ------------------------------ generic single-block exclusive scan (n<=512)
__global__ __launch_bounds__(512) void k_excl_scan(const unsigned* __restrict__ in,
                                                   unsigned* __restrict__ out, int n) {
  __shared__ unsigned buf[512];
  const int t = threadIdx.x;
  unsigned v = (t < n) ? in[t] : 0u;
  buf[t] = v;
  __syncthreads();
  for (int off = 1; off < n; off <<= 1) {
    unsigned add = (t >= off) ? buf[t - off] : 0u;
    __syncthreads();
    buf[t] += add;
    __syncthreads();
  }
  if (t < n) out[t] = buf[t] - v;
}

// ----------------------------------------- radix pass 3/3: stable scatter
__global__ __launch_bounds__(256) void k_scatter(const unsigned* __restrict__ src,
                                                 unsigned* __restrict__ dst,
                                                 const unsigned* __restrict__ hist,
                                                 const unsigned* __restrict__ dbase,
                                                 int shift) {
  __shared__ unsigned waveCnt[8][256];
  __shared__ unsigned gbase[256];
  const int t = threadIdx.x, w = t >> 5, lane = t & 31;
  const unsigned blk = blockIdx.x;
  for (int i = t; i < 2048; i += 256) (&waveCnt[0][0])[i] = 0u;
  __syncthreads();

  unsigned keys[16];
  unsigned short offs[16];
  unsigned char digs[16];
  const unsigned tileStart = blk * (unsigned)TILE;
  const unsigned ltMask = (1u << lane) - 1u;
#pragma unroll 4
  for (int j = 0; j < 16; ++j) {
    unsigned idx = ((unsigned)(w * 16 + j) * 32u + (unsigned)lane);
    unsigned key = src[tileStart + idx];
    unsigned dig = (key >> shift) & 255u;
    unsigned mask = 0xFFFFFFFFu;                  // wave32 digit-match via ballots
#pragma unroll
    for (int bb = 0; bb < 8; ++bb) {
      unsigned bal = (unsigned)__ballot((int)((dig >> bb) & 1u));
      mask &= ((dig >> bb) & 1u) ? bal : ~bal;
    }
    unsigned rank   = (unsigned)__popc(mask & ltMask);
    unsigned leader = (unsigned)(__ffs((int)mask) - 1);
    unsigned prev   = waveCnt[w][dig];
    if ((unsigned)lane == leader) waveCnt[w][dig] = prev + (unsigned)__popc(mask);
    keys[j] = key;
    digs[j] = (unsigned char)dig;
    offs[j] = (unsigned short)(prev + rank);
  }
  __syncthreads();
  {
    unsigned run = 0u;                            // cross-wave exclusive scan per digit
#pragma unroll
    for (int ww = 0; ww < 8; ++ww) { unsigned v = waveCnt[ww][t]; waveCnt[ww][t] = run; run += v; }
    gbase[t] = dbase[t] + hist[(unsigned)t * NBLK + blk];
  }
  __syncthreads();
#pragma unroll 4
  for (int j = 0; j < 16; ++j) {
    unsigned d = digs[j];
    dst[gbase[d] + waveCnt[w][d] + (unsigned)offs[j]] = keys[j];
  }
}

// -------------------------------------------------- fg counts per sorted tile
__global__ __launch_bounds__(256) void k_tile_fg(const unsigned* __restrict__ keys,
                                                 unsigned* __restrict__ tileSums) {
  __shared__ unsigned red[256];
  const int t = threadIdx.x;
  const unsigned base = blockIdx.x * (unsigned)TILE;
  unsigned s = 0u;
#pragma unroll
  for (int i = 0; i < 16; ++i) s += (~keys[base + (unsigned)i * 256u + t]) & 1u;
  red[t] = s;
  __syncthreads();
  for (int off = 128; off > 0; off >>= 1) {
    if (t < off) red[t] += red[t + off];
    __syncthreads();
  }
  if (t == 0) tileSums[blockIdx.x] = red[0];
}

// ------------------------------- Jaccard-weighted loss over the sorted array
__global__ __launch_bounds__(256) void k_tile_loss(const unsigned* __restrict__ keys,
                                                   const unsigned* __restrict__ tileBase,
                                                   const unsigned* __restrict__ Gcnt,
                                                   float* __restrict__ lossArr, int c) {
  __shared__ unsigned tsum[256];
  __shared__ float fred[256];
  const int t = threadIdx.x;
  const unsigned blk = blockIdx.x;
  const unsigned base = blk * (unsigned)TILE;
  unsigned k[16], fg[16], local = 0u;
#pragma unroll
  for (int j = 0; j < 16; ++j) {
    unsigned key = keys[base + (unsigned)t * 16u + j];
    k[j] = key;
    fg[j] = (~key) & 1u;
    local += fg[j];
  }
  tsum[t] = local;
  __syncthreads();
  for (int off = 1; off < 256; off <<= 1) {
    unsigned add = (t >= off) ? tsum[t - off] : 0u;
    __syncthreads();
    tsum[t] += add;
    __syncthreads();
  }
  unsigned F = tileBase[blk] + (tsum[t] - local);   // exclusive fg count before me
  const float Gf = (float)Gcnt[c];
  float loss = 0.0f;
#pragma unroll
  for (int j = 0; j < 16; ++j) {
    F += fg[j];                                     // inclusive cumsum F_i
    unsigned i = base + (unsigned)t * 16u + j;
    float Fi = (float)F, If = (float)i, fgf = (float)fg[j];
    float Ji   = 1.0f - (Gf - Fi) / (Gf + (If + 1.0f) - Fi);
    float Jim1 = (i == 0u) ? 0.0f
                           : 1.0f - (Gf - (Fi - fgf)) / (Gf + If - (Fi - fgf));
    float e = __uint_as_float((~k[j]) & 0xFFFFFFFEu);
    loss += e * (Ji - Jim1);
  }
  fred[t] = loss;
  __syncthreads();
  for (int off = 128; off > 0; off >>= 1) {
    if (t < off) fred[t] += fred[t + off];
    __syncthreads();
  }
  if (t == 0) atomicAdd(&lossArr[c], fred[0]);
}

// ---------------------------------------------------------------- finalize
__global__ void k_finalize(const unsigned* __restrict__ Gcnt,
                           const float* __restrict__ lossArr,
                           float* __restrict__ out) {
  if (threadIdx.x == 0) {
    float s = 0.0f, n = 0.0f;
    for (int c = 0; c < NCLS; ++c)
      if (Gcnt[c] > 0u) { s += lossArr[c]; n += 1.0f; }
    out[0] = s / fmaxf(n, 1.0f);
  }
}

// ================================================================= launch
extern "C" void kernel_launch(void* const* d_in, const int* in_sizes, int n_in,
                              void* d_out, int out_size, void* d_ws, size_t ws_size,
                              hipStream_t stream) {
  (void)in_sizes; (void)n_in; (void)out_size; (void)ws_size;
  const float* input  = (const float*)d_in[0];
  const int*   labels = (const int*)d_in[1];

  char* ws = (char*)d_ws;
  size_t off = 0;
  auto alloc = [&](size_t bytes) -> void* {
    void* p = ws + off;
    off = (off + bytes + 255u) & ~(size_t)255u;
    return p;
  };
  float*    mArr     = (float*)alloc((size_t)P_TOT * 4);
  float*    invS     = (float*)alloc((size_t)P_TOT * 4);
  unsigned* keysA    = (unsigned*)alloc((size_t)P_TOT * 4);
  unsigned* keysB    = (unsigned*)alloc((size_t)P_TOT * 4);
  unsigned* hist     = (unsigned*)alloc((size_t)256 * NBLK * 4);
  unsigned* dtot     = (unsigned*)alloc(256 * 4);
  unsigned* dbase    = (unsigned*)alloc(256 * 4);
  unsigned* tileSums = (unsigned*)alloc(NBLK * 4);
  unsigned* tileBase = (unsigned*)alloc(NBLK * 4);
  unsigned* Gcnt     = (unsigned*)alloc(NCLS * 4);
  float*    lossArr  = (float*)alloc(NCLS * 4);

  k_init<<<1, 32, 0, stream>>>(Gcnt, lossArr);
  k_softmax_stats<<<P_TOT / 256, 256, 0, stream>>>(input, mArr, invS);
  k_label_hist<<<128, 256, 0, stream>>>(labels, Gcnt);

  for (int c = 0; c < NCLS; ++c) {
    k_build_keys<<<P_TOT / 256, 256, 0, stream>>>(input, labels, mArr, invS, keysA, c);
    unsigned* src = keysA;
    unsigned* dst = keysB;
    for (int pass = 0; pass < 4; ++pass) {
      k_hist<<<NBLK, 256, 0, stream>>>(src, hist, pass * 8);
      k_scan_blocks<<<256, 512, 0, stream>>>(hist, dtot);
      k_excl_scan<<<1, 256, 0, stream>>>(dtot, dbase, 256);
      k_scatter<<<NBLK, 256, 0, stream>>>(src, dst, hist, dbase, pass * 8);
      unsigned* tmp = src; src = dst; dst = tmp;
    }
    // 4 passes -> sorted data back in keysA (== src)
    k_tile_fg<<<NBLK, 256, 0, stream>>>(src, tileSums);
    k_excl_scan<<<1, 512, 0, stream>>>(tileSums, tileBase, NBLK);
    k_tile_loss<<<NBLK, 256, 0, stream>>>(src, tileBase, Gcnt, lossArr, c);
  }
  k_finalize<<<1, 32, 0, stream>>>(Gcnt, lossArr, (float*)d_out);
}
